// GATLayer_73392401154116
// MI455X (gfx1250) — compile-verified, matrix operand order checked by reading, not run on hardware
//
#include <hip/hip_runtime.h>
#include <hip/hip_bf16.h>

#define NB   2
#define NN   4096
#define CIN  128
#define NH   4
#define CC   32      // channels per head
#define GAT_ALPHA 0.2f

typedef _Float16 v16h __attribute__((ext_vector_type(16)));
typedef _Float16 v8h  __attribute__((ext_vector_type(8)));
typedef float    v8f  __attribute__((ext_vector_type(8)));

static __device__ __forceinline__ void pack8(v16h& d, int base, float4 a, float4 b) {
    d[base + 0] = (_Float16)a.x; d[base + 1] = (_Float16)a.y;
    d[base + 2] = (_Float16)a.z; d[base + 3] = (_Float16)a.w;
    d[base + 4] = (_Float16)b.x; d[base + 5] = (_Float16)b.y;
    d[base + 6] = (_Float16)b.z; d[base + 7] = (_Float16)b.w;
}

// ---------------------------------------------------------------------------
// Phase 1: h = X @ W^T + b  (per-wave 16x32 tile via v_wmma_f32_16x16x32_f16)
// Emits: hT  (B,H,CC,N) f16   -- transposed f16 copy, B-operand-friendly
//        leftT/rightT (B,H,N) -- per-head attention scalars
// ---------------------------------------------------------------------------
__global__ __launch_bounds__(128) void gat_phase1(
    const float* __restrict__ X,      // (B,N,CIN)
    const float* __restrict__ W,      // (CIN,CIN) row-major (out,in)
    const float* __restrict__ bias,   // (CIN)
    const float* __restrict__ a,      // (NH, 2*CC)
    _Float16* __restrict__ hT,        // (B,NH,CC,N)
    float* __restrict__ leftT,        // (B,NH,N)
    float* __restrict__ rightT)       // (B,NH,N)
{
    const int blk  = blockIdx.x;
    const int bb   = blk / (NN / 16);
    const int n0   = (blk % (NN / 16)) * 16;
    const int hh   = threadIdx.x >> 5;      // wave -> head
    const int lane = threadIdx.x & 31;
    const int row  = lane & 15;
    const int half = lane >> 4;
    const int kbase = half * 8;

    const int co0 = hh * CC + row;          // output channel, col-tile 0
    const int co1 = co0 + 16;               // col-tile 1

    v8f acc0, acc1;
    const float b0 = bias[co0], b1 = bias[co1];
    #pragma unroll
    for (int r = 0; r < 8; ++r) { acc0[r] = b0; acc1[r] = b1; }

    const float* xrow = X + ((size_t)bb * NN + n0 + row) * CIN;
    const float* w0   = W + (size_t)co0 * CIN;
    const float* w1   = W + (size_t)co1 * CIN;

    #pragma unroll
    for (int kc = 0; kc < CIN; kc += 32) {
        v16h A, B0, B1;
        {
            const float4* p0 = (const float4*)(xrow + kc + kbase);
            const float4* p1 = (const float4*)(xrow + kc + kbase + 16);
            pack8(A, 0, p0[0], p0[1]);
            pack8(A, 8, p1[0], p1[1]);
        }
        {
            const float4* p0 = (const float4*)(w0 + kc + kbase);
            const float4* p1 = (const float4*)(w0 + kc + kbase + 16);
            pack8(B0, 0, p0[0], p0[1]);
            pack8(B0, 8, p1[0], p1[1]);
        }
        {
            const float4* p0 = (const float4*)(w1 + kc + kbase);
            const float4* p1 = (const float4*)(w1 + kc + kbase + 16);
            pack8(B1, 0, p0[0], p0[1]);
            pack8(B1, 8, p1[0], p1[1]);
        }
        acc0 = __builtin_amdgcn_wmma_f32_16x16x32_f16(false, A, false, B0,
                                                      (short)0, acc0, false, false);
        acc1 = __builtin_amdgcn_wmma_f32_16x16x32_f16(false, A, false, B1,
                                                      (short)0, acc1, false, false);
    }

    // Store transposed f16 h:  hT[b][h][c][n]
    _Float16* h0 = hT + (((size_t)bb * NH + hh) * CC + row) * NN;
    _Float16* h1 = hT + (((size_t)bb * NH + hh) * CC + 16 + row) * NN;
    #pragma unroll
    for (int r = 0; r < 8; ++r) {
        const int n = n0 + r + 8 * half;
        h0[n] = (_Float16)acc0[r];
        h1[n] = (_Float16)acc1[r];
    }

    // left/right: per-head dot of h-row with a_src / a_dst
    const float la0 = a[hh * 2 * CC + row];
    const float la1 = a[hh * 2 * CC + 16 + row];
    const float ld0 = a[hh * 2 * CC + CC + row];
    const float ld1 = a[hh * 2 * CC + CC + 16 + row];
    float pl[8], pr[8];
    #pragma unroll
    for (int r = 0; r < 8; ++r) {
        pl[r] = acc0[r] * la0 + acc1[r] * la1;
        pr[r] = acc0[r] * ld0 + acc1[r] * ld1;
    }
    #pragma unroll
    for (int off = 8; off; off >>= 1) {
        #pragma unroll
        for (int r = 0; r < 8; ++r) {
            pl[r] += __shfl_xor(pl[r], off, 32);
            pr[r] += __shfl_xor(pr[r], off, 32);
        }
    }
    if (row == 0) {
        float* lt = leftT  + ((size_t)bb * NH + hh) * NN;
        float* rt = rightT + ((size_t)bb * NH + hh) * NN;
        #pragma unroll
        for (int r = 0; r < 8; ++r) {
            const int n = n0 + r + 8 * half;
            lt[n] = pl[r];
            rt[n] = pr[r];
        }
    }
}

// ---------------------------------------------------------------------------
// Per-(b,head) max over rightT -> rmax (upper bound for stable softmax)
// ---------------------------------------------------------------------------
__global__ __launch_bounds__(256) void gat_rmax(const float* __restrict__ rightT,
                                                float* __restrict__ rmax)
{
    __shared__ float s[256];
    const int bh = blockIdx.x;
    const float* rt = rightT + (size_t)bh * NN;
    float m = -3.0e38f;
    for (int i = threadIdx.x; i < NN; i += 256) m = fmaxf(m, rt[i]);
    s[threadIdx.x] = m;
    __syncthreads();
    for (int o = 128; o; o >>= 1) {
        if (threadIdx.x < o) s[threadIdx.x] = fmaxf(s[threadIdx.x], s[threadIdx.x + o]);
        __syncthreads();
    }
    if (threadIdx.x == 0) rmax[bh] = s[0];
}

// ---------------------------------------------------------------------------
// Phase 2: single pass over adj.  One wave = one head; 16 i-rows per block.
// p = adj ? exp(lrelu(left+right) - bound) : 0, built directly in the
// f16 A-operand lane layout; out += P(16x32) @ hT(32x32) via 2 WMMAs/chunk.
// ---------------------------------------------------------------------------
__global__ __launch_bounds__(128) void gat_phase2(
    const int* __restrict__ adj,          // (B,N,N)
    const _Float16* __restrict__ hT,      // (B,NH,CC,N)
    const float* __restrict__ leftT,      // (B,NH,N)
    const float* __restrict__ rightT,     // (B,NH,N)
    const float* __restrict__ rmax,       // (B*NH)
    float* __restrict__ out)              // (B,N,CIN)
{
    __shared__ float ssum[NH * 16];
    const int blk  = blockIdx.x;
    const int bb   = blk / (NN / 16);
    const int i0   = (blk % (NN / 16)) * 16;
    const int hh   = threadIdx.x >> 5;
    const int lane = threadIdx.x & 31;
    const int row  = lane & 15;
    const int half = lane >> 4;
    const int kbase = half * 8;

    const float lft = leftT[((size_t)bb * NH + hh) * NN + i0 + row];
    float bnd = lft + rmax[bb * NH + hh];
    bnd = bnd > 0.f ? bnd : GAT_ALPHA * bnd;       // lrelu(left + max_j right) >= row max

    const int*      arow = adj    + ((size_t)bb * NN + i0 + row) * NN + kbase;
    const float*    rrow = rightT + ((size_t)bb * NH + hh) * NN + kbase;
    const _Float16* hb0  = hT + (((size_t)bb * NH + hh) * CC + row) * NN + kbase;
    const _Float16* hb1  = hT + (((size_t)bb * NH + hh) * CC + 16 + row) * NN + kbase;

    v8f acc0 = {}, acc1 = {};
    float rsum = 0.f;

    for (int jc = 0; jc < NN; jc += 32) {
        const int4 a0 = *(const int4*)(arow + jc);
        const int4 a1 = *(const int4*)(arow + jc + 4);
        const int4 a2 = *(const int4*)(arow + jc + 16);
        const int4 a3 = *(const int4*)(arow + jc + 20);
        const float4 r0 = *(const float4*)(rrow + jc);
        const float4 r1 = *(const float4*)(rrow + jc + 4);
        const float4 r2 = *(const float4*)(rrow + jc + 16);
        const float4 r3 = *(const float4*)(rrow + jc + 20);

        v16h A;
        {
            float pv[16];
            const int   av[16] = { a0.x,a0.y,a0.z,a0.w, a1.x,a1.y,a1.z,a1.w,
                                   a2.x,a2.y,a2.z,a2.w, a3.x,a3.y,a3.z,a3.w };
            const float rv[16] = { r0.x,r0.y,r0.z,r0.w, r1.x,r1.y,r1.z,r1.w,
                                   r2.x,r2.y,r2.z,r2.w, r3.x,r3.y,r3.z,r3.w };
            #pragma unroll
            for (int t = 0; t < 16; ++t) {
                float s = lft + rv[t];
                s = s > 0.f ? s : GAT_ALPHA * s;
                const float p = (av[t] == 1) ? __expf(s - bnd) : 0.f;
                rsum += p;
                pv[t] = p;
            }
            #pragma unroll
            for (int t = 0; t < 16; ++t) A[t] = (_Float16)pv[t];
        }

        const v8h bl00 = *(const v8h*)(hb0 + jc);
        const v8h bl01 = *(const v8h*)(hb0 + jc + 16);
        const v8h bl10 = *(const v8h*)(hb1 + jc);
        const v8h bl11 = *(const v8h*)(hb1 + jc + 16);
        const v16h B0 = __builtin_shufflevector(bl00, bl01,
                          0,1,2,3,4,5,6,7,8,9,10,11,12,13,14,15);
        const v16h B1 = __builtin_shufflevector(bl10, bl11,
                          0,1,2,3,4,5,6,7,8,9,10,11,12,13,14,15);

        acc0 = __builtin_amdgcn_wmma_f32_16x16x32_f16(false, A, false, B0,
                                                      (short)0, acc0, false, false);
        acc1 = __builtin_amdgcn_wmma_f32_16x16x32_f16(false, A, false, B1,
                                                      (short)0, acc1, false, false);
    }

    // full row sums: lane L and L+16 hold disjoint j-halves of row (L%16)
    rsum += __shfl_xor(rsum, 16, 32);
    if (lane < 16) ssum[hh * 16 + row] = rsum;
    __syncthreads();

    #pragma unroll
    for (int r = 0; r < 8; ++r) {
        const float inv = 1.0f / ssum[hh * 16 + r + 8 * half];
        const int i = i0 + r + 8 * half;
        float* op = out + ((size_t)bb * NN + i) * CIN + hh * CC;
        op[row]      = acc0[r] * inv;
        op[16 + row] = acc1[r] * inv;
    }
}

// ---------------------------------------------------------------------------
extern "C" void kernel_launch(void* const* d_in, const int* in_sizes, int n_in,
                              void* d_out, int out_size, void* d_ws, size_t ws_size,
                              hipStream_t stream) {
    const float* X    = (const float*)d_in[0];   // node_feats (B,N,128)
    const int*   adj  = (const int*)  d_in[1];   // adj_matrix (B,N,N)
    const float* W    = (const float*)d_in[2];   // (128,128)
    const float* bias = (const float*)d_in[3];   // (128,)
    const float* a    = (const float*)d_in[4];   // (4,64)
    float*       out  = (float*)d_out;

    char* ws = (char*)d_ws;
    _Float16* hT   = (_Float16*)ws;                                   // 2 MB
    float* leftT   = (float*)(ws + (size_t)NB * NH * CC * NN * 2);    // 128 KB
    float* rightT  = leftT + (size_t)NB * NH * NN;                    // 128 KB
    float* rmax    = rightT + (size_t)NB * NH * NN;                   // 32 B

    gat_phase1<<<NB * NN / 16, 128, 0, stream>>>(X, W, bias, a, hT, leftT, rightT);
    gat_rmax  <<<NB * NH,      256, 0, stream>>>(rightT, rmax);
    gat_phase2<<<NB * NN / 16, 128, 0, stream>>>(adj, hT, leftT, rightT, rmax, out);
}
